// LocalAttention_64682207477864
// MI455X (gfx1250) — matrix operand order repaired and checked by reading, not compiled, not run on hardware
//
#include <hip/hip_runtime.h>
#include <hip/hip_bf16.h>
#include <stdint.h>

#define B_    4
#define SEQ_  4096
#define D_    1024
#define SPLIT_ 4
#define L_    1024   // SEQ_/SPLIT_
#define H_    16
#define DK_   64

typedef __attribute__((ext_vector_type(16))) __bf16 v16bf;
typedef __attribute__((ext_vector_type(8)))  float  v8f;

union FragB16 { v16bf v; uint4 q[2]; unsigned short us[16]; };

__device__ __forceinline__ unsigned short f2bf(float f) {
  unsigned int u = __float_as_uint(f);
  u += 0x7FFFu + ((u >> 16) & 1u);      // round-to-nearest-even
  return (unsigned short)(u >> 16);
}

// ---------------------------------------------------------------- conversion
__global__ void cvt_f32_bf16(const float* __restrict__ in,
                             unsigned short* __restrict__ out, int n) {
  int i = blockIdx.x * blockDim.x + threadIdx.x;
  int stride = gridDim.x * blockDim.x;
  for (; i < n; i += stride) out[i] = f2bf(in[i]);
}

// ---------------------------------------------------------------- WMMA GEMM
// C[m,n] = (sum_k A[m,k] * W[k,n] + bias[n]) * scale    per z = b*SPLIT+s
// A  : bf16 [Z][L][D] rowmajor          W : bf16 [SPLIT][D][D] rowmajor
// MODE 0: bf16 out [Z][L][D]   MODE 1: bf16 out transposed [Z][D][L]
// MODE 2: f32  out [Z][L][D]
// Double-buffered LDS; A tile streamed with async global->LDS DMA (ASYNCcnt)
// overlapping the WMMA block of the previous tile.
template <int MODE>
__global__ __launch_bounds__(256) void gemm_bf16_wmma(
    const unsigned short* __restrict__ A, const unsigned short* __restrict__ W,
    const float* __restrict__ bias, void* __restrict__ out, float scale) {
  __shared__ unsigned short As[2][128 * 32];   // [buf][m][k]
  __shared__ unsigned short Bt[2][64 * 32];    // [buf][n][k] (transposed)

  const int z    = blockIdx.z;
  const int s    = z & (SPLIT_ - 1);
  const int m0   = blockIdx.y * 128;
  const int n0   = blockIdx.x * 64;
  const size_t aOff = (size_t)z * L_ * D_;
  const size_t wOff = (size_t)s * D_ * D_;
  const int    bOff = s * D_;

  const int tid   = threadIdx.x;
  const int lane  = tid & 31;
  const int wid   = tid >> 5;
  const int waveM = (wid & 3) * 32;
  const int waveN = (wid >> 2) * 32;
  const int half  = lane >> 4;
  const int l16   = lane & 15;

  // fixed per-thread staging coordinates
  const int arow = tid >> 2, acg = (tid & 3) * 8;   // A: rows arow, arow+64
  const int krow = tid >> 3, wcg = (tid & 7) * 8;   // W: one 16B chunk

  v8f acc[2][2] = {};
  const int NT = D_ / 32;

  // ---- prologue: stage tile 0 into buffer 0
#pragma unroll
  for (int i = 0; i < 2; ++i) {
    int row = arow + i * 64;
    const unsigned short* gsrc = A + aOff + (size_t)(m0 + row) * D_ + acg;
    unsigned ldsoff = (unsigned)(uintptr_t)(&As[0][row * 32 + acg]);
    asm volatile("global_load_async_to_lds_b128 %0, %1, off"
                 :: "v"(ldsoff), "v"(gsrc) : "memory");
  }
  {
    uint4 v = *(const uint4*)(W + wOff + (size_t)krow * D_ + n0 + wcg);
    unsigned int wd[4] = {v.x, v.y, v.z, v.w};
#pragma unroll
    for (int j = 0; j < 8; ++j)
      Bt[0][(wcg + j) * 32 + krow] = (unsigned short)(wd[j >> 1] >> ((j & 1) * 16));
  }
  asm volatile("s_wait_asynccnt 0" ::: "memory");
  __syncthreads();

#pragma unroll
  for (int t = 0; t < NT; ++t) {
    const int cur = t & 1, nxt = cur ^ 1;

    // ---- stage tile t+1 into the other buffer (overlaps WMMAs below)
    if (t + 1 < NT) {
      const int kk = (t + 1) * 32;
      if (t + 2 < NT) {   // warm L2/WGP$ one more tile ahead
        __builtin_prefetch(A + aOff + (size_t)(m0 + arow) * D_ + (t + 2) * 32, 0, 1);
        __builtin_prefetch(W + wOff + (size_t)((t + 2) * 32 + krow) * D_ + n0, 0, 1);
      }
#pragma unroll
      for (int i = 0; i < 2; ++i) {
        int row = arow + i * 64;
        const unsigned short* gsrc = A + aOff + (size_t)(m0 + row) * D_ + kk + acg;
        unsigned ldsoff = (unsigned)(uintptr_t)(&As[nxt][row * 32 + acg]);
        asm volatile("global_load_async_to_lds_b128 %0, %1, off"
                     :: "v"(ldsoff), "v"(gsrc) : "memory");
      }
      uint4 v = *(const uint4*)(W + wOff + (size_t)(kk + krow) * D_ + n0 + wcg);
      unsigned int wd[4] = {v.x, v.y, v.z, v.w};
#pragma unroll
      for (int j = 0; j < 8; ++j)
        Bt[nxt][(wcg + j) * 32 + krow] =
            (unsigned short)(wd[j >> 1] >> ((j & 1) * 16));
    }

    // ---- compute tile t from buffer cur
    FragB16 a[2], b[2];
#pragma unroll
    for (int mi = 0; mi < 2; ++mi) {
      int ml = waveM + mi * 16 + l16;
      a[mi].q[0] = *(const uint4*)(&As[cur][ml * 32 + half * 8]);
      a[mi].q[1] = *(const uint4*)(&As[cur][ml * 32 + 16 + half * 8]);
    }
#pragma unroll
    for (int ni = 0; ni < 2; ++ni) {
      int nl = waveN + ni * 16 + l16;
      b[ni].q[0] = *(const uint4*)(&Bt[cur][nl * 32 + half * 8]);
      b[ni].q[1] = *(const uint4*)(&Bt[cur][nl * 32 + 16 + half * 8]);
    }
#pragma unroll
    for (int mi = 0; mi < 2; ++mi)
#pragma unroll
      for (int ni = 0; ni < 2; ++ni)
        acc[mi][ni] = __builtin_amdgcn_wmma_f32_16x16x32_bf16(
            false, a[mi].v, false, b[ni].v, (short)0, acc[mi][ni], false, false);

    // DMA for t+1 must be complete (and every wave past its reads of cur)
    asm volatile("s_wait_asynccnt 0" ::: "memory");
    __syncthreads();
  }

  // epilogue: bias + scale, per C layout (row = half*8+r, col = l16)
#pragma unroll
  for (int mi = 0; mi < 2; ++mi) {
#pragma unroll
    for (int ni = 0; ni < 2; ++ni) {
      int ng = n0 + waveN + ni * 16 + l16;
      float bvv = bias[bOff + ng];
#pragma unroll
      for (int r = 0; r < 8; ++r) {
        int mg = m0 + waveM + mi * 16 + half * 8 + r;
        float val = (acc[mi][ni][r] + bvv) * scale;
        if (MODE == 0)
          ((unsigned short*)out)[aOff + (size_t)mg * D_ + ng] = f2bf(val);
        else if (MODE == 1)
          ((unsigned short*)out)[aOff + (size_t)ng * L_ + mg] = f2bf(val);
        else
          ((float*)out)[aOff + (size_t)mg * D_ + ng] = val;
      }
    }
  }
}

// ------------------------------------------------------------- attention
// Per (z, head): flash-style causal attention within one L=1024 block.
// Q : [Z][L][D] bf16 (pre-scaled by 1/8)   K : [Z][L][D] bf16
// Vt: [Z][D][L] bf16 (transposed)          O : [Z][L][D] bf16
__global__ __launch_bounds__(256) void attn_wmma(
    const unsigned short* __restrict__ Q, const unsigned short* __restrict__ K,
    const unsigned short* __restrict__ Vt, unsigned short* __restrict__ O) {
  __shared__ unsigned short Pt[8][16 * 32];   // wave-private P tiles

  const int z    = blockIdx.z;
  const int h    = blockIdx.y;
  const int tid  = threadIdx.x;
  const int lane = tid & 31;
  const int wid  = tid >> 5;
  const int half = lane >> 4;
  const int l16  = lane & 15;
  const int mw0  = blockIdx.x * 128 + wid * 16;   // this wave's 16 query rows

  const size_t base  = (size_t)z * L_ * D_;
  const size_t baseT = (size_t)z * D_ * L_;
  const int hc = h * DK_;

  // Q A-fragments for both 32-wide k-chunks (DK=64)
  FragB16 qa[2];
  {
    const unsigned short* qrow = Q + base + (size_t)(mw0 + l16) * D_ + hc;
#pragma unroll
    for (int kc = 0; kc < 2; ++kc) {
      qa[kc].q[0] = *(const uint4*)(qrow + kc * 32 + half * 8);
      qa[kc].q[1] = *(const uint4*)(qrow + kc * 32 + 16 + half * 8);
    }
  }

  v8f o[4] = {};
  float rmax[8], rsum[8];
#pragma unroll
  for (int r = 0; r < 8; ++r) { rmax[r] = -3.0e38f; rsum[r] = 0.0f; }
  unsigned short* myP = Pt[wid];

  const int nblocks = (mw0 + 16 + 31) >> 5;       // causal trip count
  for (int jb = 0; jb < nblocks; ++jb) {
    const int j0 = jb * 32;

    // S = Q K^T : two 16-key subtiles, K rows contiguous in k-dim
    v8f sc[2] = {};
#pragma unroll
    for (int sub = 0; sub < 2; ++sub) {
      const unsigned short* krow =
          K + base + (size_t)(j0 + sub * 16 + l16) * D_ + hc;
#pragma unroll
      for (int kc = 0; kc < 2; ++kc) {
        FragB16 bk;
        bk.q[0] = *(const uint4*)(krow + kc * 32 + half * 8);
        bk.q[1] = *(const uint4*)(krow + kc * 32 + 16 + half * 8);
        sc[sub] = __builtin_amdgcn_wmma_f32_16x16x32_bf16(
            false, qa[kc].v, false, bk.v, (short)0, sc[sub], false, false);
      }
    }

    // causal mask (C layout: key = j0+sub*16+l16, row = mw0+half*8+r)
#pragma unroll
    for (int sub = 0; sub < 2; ++sub) {
      int key = j0 + sub * 16 + l16;
#pragma unroll
      for (int r = 0; r < 8; ++r)
        if (key > mw0 + half * 8 + r) sc[sub][r] = -1.0e30f;
    }

    // online softmax per row; stash P (bf16) into wave-private LDS
#pragma unroll
    for (int r = 0; r < 8; ++r) {
      float mx = fmaxf(sc[0][r], sc[1][r]);
      mx = fmaxf(mx, __shfl_xor(mx, 1, 32));
      mx = fmaxf(mx, __shfl_xor(mx, 2, 32));
      mx = fmaxf(mx, __shfl_xor(mx, 4, 32));
      mx = fmaxf(mx, __shfl_xor(mx, 8, 32));
      float nm   = fmaxf(rmax[r], mx);
      float corr = __expf(rmax[r] - nm);
      rmax[r] = nm;
      float p0 = __expf(sc[0][r] - nm);
      float p1 = __expf(sc[1][r] - nm);
      float ls = p0 + p1;
      ls += __shfl_xor(ls, 1, 32);
      ls += __shfl_xor(ls, 2, 32);
      ls += __shfl_xor(ls, 4, 32);
      ls += __shfl_xor(ls, 8, 32);
      rsum[r] = rsum[r] * corr + ls;
#pragma unroll
      for (int t = 0; t < 4; ++t) o[t][r] *= corr;
      int ml = half * 8 + r;
      myP[ml * 32 + l16]      = f2bf(p0);
      myP[ml * 32 + 16 + l16] = f2bf(p1);
    }
    // wave-private LDS round trip: wait on split DS counter (CDNA5)
    asm volatile("s_wait_dscnt 0" ::: "memory");

    FragB16 pa;
    pa.q[0] = *(const uint4*)(&myP[l16 * 32 + half * 8]);
    pa.q[1] = *(const uint4*)(&myP[l16 * 32 + 16 + half * 8]);

    // O += P @ V : Vt columns contiguous over keys
#pragma unroll
    for (int t = 0; t < 4; ++t) {
      const unsigned short* vcol =
          Vt + baseT + (size_t)(hc + t * 16 + l16) * L_ + j0;
      FragB16 bv;
      bv.q[0] = *(const uint4*)(vcol + half * 8);
      bv.q[1] = *(const uint4*)(vcol + 16 + half * 8);
      o[t] = __builtin_amdgcn_wmma_f32_16x16x32_bf16(
          false, pa.v, false, bv.v, (short)0, o[t], false, false);
    }
  }

  // normalize and store O (bf16)
  unsigned short* obase = O + base + hc;
#pragma unroll
  for (int r = 0; r < 8; ++r) {
    float inv = 1.0f / rsum[r];
    size_t mrow = (size_t)(mw0 + half * 8 + r) * D_;
#pragma unroll
    for (int t = 0; t < 4; ++t)
      obase[mrow + t * 16 + l16] = f2bf(o[t][r] * inv);
  }
}

// ---------------------------------------------------------------- launch
extern "C" void kernel_launch(void* const* d_in, const int* in_sizes, int n_in,
                              void* d_out, int out_size, void* d_ws, size_t ws_size,
                              hipStream_t stream) {
  (void)in_sizes; (void)n_in; (void)out_size; (void)ws_size;
  const float* x  = (const float*)d_in[0];
  const float* Wq = (const float*)d_in[1];
  const float* Wk = (const float*)d_in[2];
  const float* Wv = (const float*)d_in[3];
  const float* Wo = (const float*)d_in[4];
  const float* bq = (const float*)d_in[5];
  const float* bk = (const float*)d_in[6];
  const float* bv = (const float*)d_in[7];
  const float* bo = (const float*)d_in[8];

  char* ws = (char*)d_ws;
  const size_t MB = 1u << 20;
  unsigned short* xb  = (unsigned short*)(ws + 0);          // 32 MB
  unsigned short* wqb = (unsigned short*)(ws + 32 * MB);    //  8 MB
  unsigned short* wkb = (unsigned short*)(ws + 40 * MB);
  unsigned short* wvb = (unsigned short*)(ws + 48 * MB);
  unsigned short* wob = (unsigned short*)(ws + 56 * MB);
  unsigned short* qb  = (unsigned short*)(ws + 64 * MB);    // 32 MB each
  unsigned short* kb  = (unsigned short*)(ws + 96 * MB);
  unsigned short* vtb = (unsigned short*)(ws + 128 * MB);   // V transposed [Z][D][L]
  unsigned short* ob  = (unsigned short*)(ws + 160 * MB);

  const int NX = B_ * SEQ_ * D_;
  const int NW = SPLIT_ * D_ * D_;
  cvt_f32_bf16<<<2048, 256, 0, stream>>>(x,  xb,  NX);
  cvt_f32_bf16<<<1024, 256, 0, stream>>>(Wq, wqb, NW);
  cvt_f32_bf16<<<1024, 256, 0, stream>>>(Wk, wkb, NW);
  cvt_f32_bf16<<<1024, 256, 0, stream>>>(Wv, wvb, NW);
  cvt_f32_bf16<<<1024, 256, 0, stream>>>(Wo, wob, NW);

  dim3 blk(256);
  dim3 grd(D_ / 64, L_ / 128, B_ * SPLIT_);   // 16 x 8 x 16
  gemm_bf16_wmma<0><<<grd, blk, 0, stream>>>(xb, wqb, bq, qb, 0.125f); // Q * 1/sqrt(DK)
  gemm_bf16_wmma<0><<<grd, blk, 0, stream>>>(xb, wkb, bk, kb, 1.0f);
  gemm_bf16_wmma<1><<<grd, blk, 0, stream>>>(xb, wvb, bv, vtb, 1.0f); // V transposed

  dim3 agrd(L_ / 128, H_, B_ * SPLIT_);       // 8 x 16 x 16
  attn_wmma<<<agrd, blk, 0, stream>>>(qb, kb, vtb, ob);

  gemm_bf16_wmma<2><<<grd, blk, 0, stream>>>(ob, wob, bo, d_out, 1.0f);
}
